// BiFormerBlock_25855703122692
// MI455X (gfx1250) — matrix operand order repaired and checked by reading, not compiled, not run on hardware
//
#include <hip/hip_runtime.h>
#include <hip/hip_bf16.h>

typedef __attribute__((ext_vector_type(16))) _Float16 v16h;
typedef __attribute__((ext_vector_type(8)))  float    v8f;

#define TOK   25088   // 8*56*56
#define DIM   256
#define QKD   256
#define HEADS 8
#define DQ    32
#define W2    64      // tokens per window
#define P2    49      // windows per image
#define NB    8       // batch
#define SCALEQ 0.0625f // 256^-0.5

// ---------------------------------------------------------------------------
// Tensor Data Mover (gfx1250) — this toolchain exposes the 6-arg builtin:
//   (uint32x4 g0, int32x8 g1, int32x4 g2, int32x4 g3, int32x8 g4, i32 cpol)
#if defined(__has_builtin)
#if __has_builtin(__builtin_amdgcn_tensor_load_to_lds)
#define HAVE_TDM 1
#endif
#endif
#ifndef HAVE_TDM
#define HAVE_TDM 0
#endif

#if HAVE_TDM
typedef unsigned int tdm_v4u __attribute__((ext_vector_type(4)));
typedef int          tdm_v8i __attribute__((ext_vector_type(8)));
typedef int          tdm_v4i __attribute__((ext_vector_type(4)));

// 2D tile load: tile_rows rows of row_u8*8 bytes, row stride stride_u8*8 bytes.
// lds_byte_addr = workgroup-relative LDS byte address (= low 32 bits of flat addr).
__device__ __forceinline__ void tdm_load_2d(unsigned lds_byte_addr, const void* gaddr,
                                            unsigned tile_rows, unsigned row_u8,
                                            unsigned stride_u8)
{
    unsigned long long ga = (unsigned long long)(size_t)gaddr;
    tdm_v4u g0;
    g0[0] = 1u;                                  // count=1 (valid), user descriptor
    g0[1] = lds_byte_addr;                       // lds_addr  (bits 63:32)
    g0[2] = (unsigned)ga;                        // global_addr[31:0]
    g0[3] = (unsigned)(ga >> 32) | (2u << 30);   // global_addr[56:32] | type=2 ("image")
    unsigned td0 = stride_u8;                    // tensor_dim0 (8B units)
    unsigned td1 = tile_rows;                    // tensor_dim1
    tdm_v8i g1;
    g1[0] = (int)(3u << 16);                                  // data_size=3 (8B)
    g1[1] = (int)((td0 & 0xFFFFu) << 16);                     // tensor_dim0[15:0]
    g1[2] = (int)((td0 >> 16) | ((td1 & 0xFFFFu) << 16));     // dim0[31:16] | dim1[15:0]
    g1[3] = (int)((td1 >> 16) | (row_u8 << 16));              // dim1[31:16] | tile_dim0
    g1[4] = (int)tile_rows;                                   // tile_dim1 (tile_dim2=0)
    g1[5] = (int)stride_u8;                                   // tensor_dim0_stride[31:0]
    g1[6] = 0;                                                // stride[47:32], dim1_stride lo
    g1[7] = 0;
    tdm_v4i z4 = {0, 0, 0, 0};
    tdm_v8i z8 = {0, 0, 0, 0, 0, 0, 0, 0};
    __builtin_amdgcn_tensor_load_to_lds(g0, g1, z4, z4, z8, 0);
}
#endif

// ---------------------------------------------------------------------------
// WMMA fragment loaders (wave32, 16x16x32 f16)
__device__ __forceinline__ v16h frag_ldA(const _Float16* tile, int ldk) {
    int lane = threadIdx.x & 31;
    int r = lane & 15;
    int h = (lane >> 4) * 8;
    const _Float16* p = tile + r * ldk + h;
    v16h f;
    ((uint4*)&f)[0] = *(const uint4*)(p);
    ((uint4*)&f)[1] = *(const uint4*)(p + 16);
    return f;
}
__device__ __forceinline__ v16h frag_ldB(const _Float16* tile, int ldk) {
    int lane = threadIdx.x & 31;
    int nn = lane & 15;
    int h = (lane >> 4) * 8;
    const _Float16* p = tile + nn * ldk + h;
    v16h f;
    ((uint4*)&f)[0] = *(const uint4*)(p);
    ((uint4*)&f)[1] = *(const uint4*)(p + 16);
    return f;
}
#define WMMA_F32_F16(a, b, c) \
    __builtin_amdgcn_wmma_f32_16x16x32_f16(false, (a), false, (b), (short)0, (c), false, false)

// ---------------------------------------------------------------------------
// Generic tiled GEMM: C[MxN] = A16[MxK] * Bt16[NxK] + bias, epilogue modes.
#define GM_F16      0
#define GM_F32_RES  1
#define GM_F16_GELU 2

__global__ __launch_bounds__(128)
void gemm_wmma(const _Float16* __restrict__ A, const _Float16* __restrict__ Bt,
               const float* __restrict__ bias, const float* __restrict__ residual,
               void* __restrict__ Cout, int M, int N, int K, int mode)
{
    __shared__ __align__(16) _Float16 As[2][64][32];
    __shared__ __align__(16) _Float16 Bs[2][64][32];

    int tid  = threadIdx.x;
    int wave = tid >> 5, lane = tid & 31;
    int wr = (wave >> 1) * 32;
    int wc = (wave & 1) * 32;
    long brow = (long)blockIdx.y * 64;
    long bcol = (long)blockIdx.x * 64;

    v8f acc[2][2] = {};
    int ktiles = K >> 5;

#if HAVE_TDM
    // ---- TDM double-buffered pipeline: DMA tile t+1 while WMMAs consume tile t
    unsigned ldsA0 = (unsigned)(size_t)&As[0][0][0];
    unsigned ldsA1 = (unsigned)(size_t)&As[1][0][0];
    unsigned ldsB0 = (unsigned)(size_t)&Bs[0][0][0];
    unsigned ldsB1 = (unsigned)(size_t)&Bs[1][0][0];
    unsigned su = (unsigned)(K >> 2);                 // row stride, 8B units (K*2/8)
    const _Float16* Abase = A  + brow * (long)K;
    const _Float16* Bbase = Bt + bcol * (long)K;

    if (wave == 0) {
        tdm_load_2d(ldsA0, Abase, 64, 8, su);
        tdm_load_2d(ldsB0, Bbase, 64, 8, su);
    }
    for (int t = 0; t < ktiles; ++t) {
        if (wave == 0) __builtin_amdgcn_s_wait_tensorcnt(0);   // tile t landed
        __syncthreads();                                       // visible to all waves
        if (wave == 0 && t + 1 < ktiles) {                     // prefetch tile t+1
            unsigned bsel = (unsigned)(t + 1) & 1u;
            tdm_load_2d(bsel ? ldsA1 : ldsA0, Abase + (t + 1) * 32, 64, 8, su);
            tdm_load_2d(bsel ? ldsB1 : ldsB0, Bbase + (t + 1) * 32, 64, 8, su);
        }
        int bsel = t & 1;
        v16h a0 = frag_ldA(&As[bsel][wr][0],      32);
        v16h a1 = frag_ldA(&As[bsel][wr + 16][0], 32);
        v16h b0 = frag_ldB(&Bs[bsel][wc][0],      32);
        v16h b1 = frag_ldB(&Bs[bsel][wc + 16][0], 32);
        acc[0][0] = WMMA_F32_F16(a0, b0, acc[0][0]);
        acc[0][1] = WMMA_F32_F16(a0, b1, acc[0][1]);
        acc[1][0] = WMMA_F32_F16(a1, b0, acc[1][0]);
        acc[1][1] = WMMA_F32_F16(a1, b1, acc[1][1]);
        // no trailing barrier: next iteration's wait+barrier orders buffer reuse,
        // and __syncthreads' dscnt wait retires this tile's ds_loads first.
    }
#else
    // ---- fallback: VGPR-staged (global_load_b128 + ds_store_b128)
    int r = tid >> 1, hh = tid & 1;
    for (int kt = 0; kt < K; kt += 32) {
        const uint4* sa = (const uint4*)(A  + (brow + r) * (long)K + kt) + hh * 2;
        const uint4* sb = (const uint4*)(Bt + (bcol + r) * (long)K + kt) + hh * 2;
        uint4* da = (uint4*)(&As[0][r][0]) + hh * 2;
        uint4* db = (uint4*)(&Bs[0][r][0]) + hh * 2;
        da[0] = sa[0]; da[1] = sa[1];
        db[0] = sb[0]; db[1] = sb[1];
        __syncthreads();
        v16h a0 = frag_ldA(&As[0][wr][0],      32);
        v16h a1 = frag_ldA(&As[0][wr + 16][0], 32);
        v16h b0 = frag_ldB(&Bs[0][wc][0],      32);
        v16h b1 = frag_ldB(&Bs[0][wc + 16][0], 32);
        acc[0][0] = WMMA_F32_F16(a0, b0, acc[0][0]);
        acc[0][1] = WMMA_F32_F16(a0, b1, acc[0][1]);
        acc[1][0] = WMMA_F32_F16(a1, b0, acc[1][0]);
        acc[1][1] = WMMA_F32_F16(a1, b1, acc[1][1]);
        __syncthreads();
    }
#endif

    int nn = lane & 15, mh = (lane >> 4) * 8;
    for (int i = 0; i < 2; ++i)
        for (int j = 0; j < 2; ++j) {
            long gr0 = brow + wr + i * 16 + mh;
            long gc  = bcol + wc + j * 16 + nn;
            float bv = bias ? bias[gc] : 0.0f;
            for (int e = 0; e < 8; ++e) {
                float v = acc[i][j][e] + bv;
                long idx = (gr0 + e) * (long)N + gc;
                if (mode == GM_F16) {
                    ((_Float16*)Cout)[idx] = (_Float16)v;
                } else if (mode == GM_F32_RES) {
                    ((float*)Cout)[idx] = v + residual[idx];
                } else { // exact GELU
                    float g = 0.5f * v * (1.0f + erff(v * 0.70710678118654752f));
                    ((_Float16*)Cout)[idx] = (_Float16)g;
                }
            }
        }
}

// ---------------------------------------------------------------------------
// LayerNorm (one block per token, 256 threads), optional image->window reorder
__global__ __launch_bounds__(256)
void ln_kernel(const float* __restrict__ x, const float* __restrict__ g,
               const float* __restrict__ b, _Float16* __restrict__ out, int window_reorder)
{
    __shared__ float red[256];
    long tok = blockIdx.x;
    int c = threadIdx.x;
    float v = x[tok * DIM + c];
    red[c] = v; __syncthreads();
    for (int s = 128; s > 0; s >>= 1) { if (c < s) red[c] += red[c + s]; __syncthreads(); }
    float mu = red[0] * (1.0f / DIM);
    __syncthreads();
    float d = v - mu;
    red[c] = d * d; __syncthreads();
    for (int s = 128; s > 0; s >>= 1) { if (c < s) red[c] += red[c + s]; __syncthreads(); }
    float var = red[0] * (1.0f / DIM);
    float y = d * rsqrtf(var + 1e-6f) * g[c] + b[c];
    long ot = tok;
    if (window_reorder) {
        int n = (int)(tok / 3136), rem = (int)(tok % 3136);
        int yy = rem / 56, xx = rem % 56;
        int j = yy >> 3, hrow = yy & 7, ii = xx >> 3, ww = xx & 7;
        ot = (long)(n * P2 + j * 7 + ii) * W2 + hrow * 8 + ww;
    }
    out[ot * DIM + c] = (_Float16)y;
}

// ---------------------------------------------------------------------------
// split qkv (window order), fold SCALE into q, scatter v to image layout fp32
__global__ __launch_bounds__(256)
void split_qkv(const _Float16* __restrict__ qkv, _Float16* __restrict__ q,
               _Float16* __restrict__ k, _Float16* __restrict__ v, float* __restrict__ vimg)
{
    long wt = blockIdx.x;
    int c = threadIdx.x;
    const _Float16* row = qkv + wt * 768;
    _Float16 qv = row[c], kv = row[256 + c], vv = row[512 + c];
    q[wt * DIM + c] = (_Float16)((float)qv * SCALEQ);
    k[wt * DIM + c] = kv;
    v[wt * DIM + c] = vv;
    int n = (int)(wt / 3136), rem = (int)(wt % 3136);
    int p = rem >> 6, s = rem & 63;
    int j = p / 7, ii = p % 7, hrow = s >> 3, ww = s & 7;
    long it = ((long)n * 56 + j * 8 + hrow) * 56 + ii * 8 + ww;
    vimg[it * DIM + c] = (float)vv;
}

// window means (q already scaled; mean(q*S) == mean(q)*S)
__global__ __launch_bounds__(256)
void win_mean(const _Float16* __restrict__ q, const _Float16* __restrict__ k,
              float* __restrict__ qwin, float* __restrict__ kwin)
{
    long np = blockIdx.x;
    int c = threadIdx.x;
    float sq = 0.f, sk = 0.f;
    for (int s = 0; s < W2; ++s) {
        sq += (float)q[(np * W2 + s) * DIM + c];
        sk += (float)k[(np * W2 + s) * DIM + c];
    }
    qwin[np * DIM + c] = sq * (1.0f / W2);
    kwin[np * DIM + c] = sk * (1.0f / W2);
}

// routing top-4 per (n, p)
__global__ __launch_bounds__(64)
void routing_topk(const float* __restrict__ qwin, const float* __restrict__ kwin,
                  int* __restrict__ ridx)
{
    __shared__ float lg[P2];
    int np = blockIdx.x, n = np / P2;
    int t = threadIdx.x;
    if (t < P2) {
        const float* qp = qwin + (long)np * DIM;
        const float* kp = kwin + ((long)n * P2 + t) * DIM;
        float s = 0.f;
        for (int c = 0; c < DIM; ++c) s += qp[c] * kp[c];
        lg[t] = s;
    }
    __syncthreads();
    if (t == 0) {
        for (int sel = 0; sel < 4; ++sel) {
            int best = 0; float bv = -1e30f;
            for (int q2 = 0; q2 < P2; ++q2) { float v = lg[q2]; if (v > bv) { bv = v; best = q2; } }
            ridx[np * 4 + sel] = best;
            lg[best] = -1e30f;
        }
    }
}

// ---------------------------------------------------------------------------
// Flash-style windowed attention: block per (n, p, head), 4 waves (1 q-tile each)
__global__ __launch_bounds__(128)
void bra_attention(const _Float16* __restrict__ q16, const _Float16* __restrict__ k16,
                   const _Float16* __restrict__ v16, const int* __restrict__ ridx,
                   _Float16* __restrict__ out16)
{
    __shared__ __align__(16) _Float16 qs[64][32];
    __shared__ __align__(16) _Float16 ks[64][32];
    __shared__ __align__(16) _Float16 vst[32][64];   // V^T for B operand
    __shared__ float ls[4][16][64];
    __shared__ float rowscale[4][16];
    __shared__ float rowl[4][16];

    int blk  = blockIdx.x;
    int head = blk & 7;
    int np   = blk >> 3;
    int n    = np / P2;
    int tid = threadIdx.x, wave = tid >> 5, lane = tid & 31;

    // stage q tile (64x32 f16 rows of 64B at 512B stride)
#if HAVE_TDM
    if (wave == 0) {
        tdm_load_2d((unsigned)(size_t)&qs[0][0],
                    q16 + ((long)np * W2) * QKD + head * DQ, 64, 8, QKD >> 2);
        __builtin_amdgcn_s_wait_tensorcnt(0);
    }
#else
    {
        int r = tid >> 1, h = tid & 1;
        const uint4* src = (const uint4*)(q16 + ((long)np * W2 + r) * QKD + head * DQ) + h * 2;
        uint4* dst = (uint4*)(&qs[r][0]) + h * 2;
        dst[0] = src[0]; dst[1] = src[1];
    }
#endif
    __syncthreads();
    v16h aq = frag_ldA(&qs[wave * 16][0], 32);

    v8f  acc[2] = {};
    float m_run = -1e30f, l_run = 0.0f;
    int myrow = lane >> 1;
    int cseg  = (lane & 1) * 32;

    for (int t = 0; t < 4; ++t) {
        int widx = ridx[np * 4 + t];
        long kb = ((long)(n * P2 + widx)) * W2;
#if HAVE_TDM
        if (wave == 0)
            tdm_load_2d((unsigned)(size_t)&ks[0][0],
                        k16 + kb * QKD + head * DQ, 64, 8, QKD >> 2);
        { // V^T scatter (TDM cannot transpose)
            int r = tid >> 1, h = tid & 1;
            const _Float16* vsrc = v16 + (kb + r) * DIM + head * DQ + h * 16;
            for (int d = 0; d < 16; ++d) vst[h * 16 + d][r] = vsrc[d];
        }
        if (wave == 0) __builtin_amdgcn_s_wait_tensorcnt(0);
#else
        {
            int r = tid >> 1, h = tid & 1;
            const uint4* src = (const uint4*)(k16 + (kb + r) * QKD + head * DQ) + h * 2;
            uint4* dst = (uint4*)(&ks[r][0]) + h * 2;
            dst[0] = src[0]; dst[1] = src[1];
            const _Float16* vsrc = v16 + (kb + r) * DIM + head * DQ + h * 16;
            for (int d = 0; d < 16; ++d) vst[h * 16 + d][r] = vsrc[d];
        }
#endif
        __syncthreads();

        // S = q * k^T  (16x64 per wave)
        int nn = lane & 15, mh = (lane >> 4) * 8;
        for (int nt = 0; nt < 4; ++nt) {
            v16h bk = frag_ldB(&ks[nt * 16][0], 32);
            v8f z = {};
            v8f s = WMMA_F32_F16(aq, bk, z);
            for (int e = 0; e < 8; ++e) ls[wave][mh + e][nt * 16 + nn] = s[e];
        }
        __syncthreads();

        // streaming softmax: lane owns row myrow, half-cols cseg..cseg+31
        float mx = -1e30f;
        for (int cj = 0; cj < 32; ++cj) mx = fmaxf(mx, ls[wave][myrow][cseg + cj]);
        mx = fmaxf(mx, __shfl_xor(mx, 1, 32));
        float m_new = fmaxf(m_run, mx);
        float scale = __expf(m_run - m_new);
        float sum = 0.f;
        for (int cj = 0; cj < 32; ++cj) {
            float p = __expf(ls[wave][myrow][cseg + cj] - m_new);
            ls[wave][myrow][cseg + cj] = p;
            sum += p;
        }
        sum += __shfl_xor(sum, 1, 32);
        l_run = l_run * scale + sum;
        m_run = m_new;
        if ((lane & 1) == 0) rowscale[wave][myrow] = scale;
        __syncthreads();

        // rescale accumulators
        for (int f = 0; f < 2; ++f)
            for (int e = 0; e < 8; ++e)
                acc[f][e] *= rowscale[wave][mh + e];

        // acc += P * V
        for (int kc = 0; kc < 2; ++kc) {
            v16h ap;
            {
                int rr = lane & 15, h = (lane >> 4) * 8;
                const float* s0 = &ls[wave][rr][kc * 32 + h];
                const float* s1 = &ls[wave][rr][kc * 32 + 16 + h];
                for (int e = 0; e < 8; ++e) { ap[e] = (_Float16)s0[e]; ap[8 + e] = (_Float16)s1[e]; }
            }
            for (int nt = 0; nt < 2; ++nt) {
                v16h bv = frag_ldB(&vst[nt * 16][kc * 32], 64);
                acc[nt] = WMMA_F32_F16(ap, bv, acc[nt]);
            }
        }
        __syncthreads();
    }

    if ((lane & 1) == 0) rowl[wave][myrow] = l_run;
    __syncthreads();
    {
        int nn = lane & 15, mh = (lane >> 4) * 8;
        for (int nt = 0; nt < 2; ++nt)
            for (int e = 0; e < 8; ++e) {
                int row = wave * 16 + mh + e;
                float o = acc[nt][e] / rowl[wave][mh + e];
                out16[((long)np * W2 + row) * DIM + head * DQ + nt * 16 + nn] = (_Float16)o;
            }
    }
}

// ---------------------------------------------------------------------------
// depthwise 5x5 SAME conv (lepe), fp32, image layout
__global__ __launch_bounds__(256)
void lepe_conv(const float* __restrict__ vimg, const float* __restrict__ w,
               const float* __restrict__ b, float* __restrict__ out)
{
    long it = blockIdx.x;
    int c = threadIdx.x;
    int n = (int)(it / 3136), rem = (int)(it % 3136);
    int y = rem / 56, x = rem % 56;
    float s = b[c];
    for (int dy = 0; dy < 5; ++dy) {
        int yy = y + dy - 2; if (yy < 0 || yy >= 56) continue;
        for (int dx = 0; dx < 5; ++dx) {
            int xx = x + dx - 2; if (xx < 0 || xx >= 56) continue;
            s += vimg[(((long)n * 56 + yy) * 56 + xx) * DIM + c] * w[(dy * 5 + dx) * DIM + c];
        }
    }
    out[it * DIM + c] = s;
}

// window->image reorder + lepe add, f16 out for wo GEMM
__global__ __launch_bounds__(256)
void add_lepe_reorder(const _Float16* __restrict__ attn, const float* __restrict__ lepe,
                      _Float16* __restrict__ a16)
{
    long wt = blockIdx.x;
    int c = threadIdx.x;
    int n = (int)(wt / 3136), rem = (int)(wt % 3136);
    int p = rem >> 6, s = rem & 63;
    int j = p / 7, ii = p % 7, hrow = s >> 3, ww = s & 7;
    long it = ((long)n * 56 + j * 8 + hrow) * 56 + ii * 8 + ww;
    a16[it * DIM + c] = (_Float16)((float)attn[wt * DIM + c] + lepe[it * DIM + c]);
}

// fp32 [K][N] weight -> f16 transposed [N][K]
__global__ __launch_bounds__(256)
void conv_w_t(const float* __restrict__ w, _Float16* __restrict__ wt, int K, int N)
{
    long i = (long)blockIdx.x * 256 + threadIdx.x;
    if (i >= (long)K * N) return;
    int k = (int)(i / N), n2 = (int)(i % N);
    wt[(long)n2 * K + k] = (_Float16)w[i];
}

// ---------------------------------------------------------------------------
extern "C" void kernel_launch(void* const* d_in, const int* in_sizes, int n_in,
                              void* d_out, int out_size, void* d_ws, size_t ws_size,
                              hipStream_t stream)
{
    (void)in_sizes; (void)n_in; (void)out_size; (void)ws_size;
    const float* x      = (const float*)d_in[0];
    const float* ln1_g  = (const float*)d_in[1];
    const float* ln1_b  = (const float*)d_in[2];
    const float* qkv_w  = (const float*)d_in[3];
    const float* qkv_b  = (const float*)d_in[4];
    const float* lepe_w = (const float*)d_in[5];
    const float* lepe_b = (const float*)d_in[6];
    const float* wo_w   = (const float*)d_in[7];
    const float* wo_b   = (const float*)d_in[8];
    const float* ln2_g  = (const float*)d_in[9];
    const float* ln2_b  = (const float*)d_in[10];
    const float* mlp_w1 = (const float*)d_in[11];
    const float* mlp_b1 = (const float*)d_in[12];
    const float* mlp_w2 = (const float*)d_in[13];
    const float* mlp_b2 = (const float*)d_in[14];

    char* ws = (char*)d_ws;
    const size_t MB = 1024 * 1024;
    // lifetime-aliased workspace carve (total ~193 MB)
    _Float16* xn16   = (_Float16*)(ws + 0);          // 12.85MB; dead after qkv GEMM
    _Float16* qkv16  = (_Float16*)(ws + 16 * MB);    // 38.6MB;  dead after split
    _Float16* q16    = (_Float16*)(ws + 56 * MB);    // 12.85MB; dead after attention
    _Float16* k16    = (_Float16*)(ws + 72 * MB);
    _Float16* v16    = (_Float16*)(ws + 88 * MB);
    float*    v32    = (float*)   (ws + 104 * MB);   // 25.7MB;  dead after lepe
    float*    lepe32 = (float*)   (ws + 132 * MB);   // 25.7MB
    float*    x1     = (float*)   (ws + 160 * MB);   // attn-block output (residual stream)
    float*    qwin   = (float*)   (ws + 186 * MB);
    float*    kwin   = (float*)   (ws + 187 * MB);
    int*      ridx   = (int*)     (ws + 188 * MB);
    _Float16* wqkv_t = (_Float16*)(ws + 189 * MB);
    _Float16* wo_t   = (_Float16*)(ws + 190 * MB);
    _Float16* w1_t   = (_Float16*)(ws + 191 * MB);
    _Float16* w2_t   = (_Float16*)(ws + 192 * MB);
    _Float16* attn16 = xn16;                          // alias (xn16 dead by then)
    _Float16* a16    = qkv16;                         // alias (qkv16 dead by then)
    _Float16* h16    = q16;                           // alias (q16 dead by then)
    _Float16* hid16  = k16;                           // 51.4MB over k16+v16+v32 (all dead)

    // weights -> f16 transposed
    conv_w_t<<<(256 * 768 + 255) / 256, 256, 0, stream>>>(qkv_w, wqkv_t, 256, 768);
    conv_w_t<<<(256 * 256 + 255) / 256, 256, 0, stream>>>(wo_w, wo_t, 256, 256);
    conv_w_t<<<(256 * 1024 + 255) / 256, 256, 0, stream>>>(mlp_w1, w1_t, 256, 1024);
    conv_w_t<<<(1024 * 256 + 255) / 256, 256, 0, stream>>>(mlp_w2, w2_t, 1024, 256);

    // LN1 (-> window order f16)
    ln_kernel<<<TOK, 256, 0, stream>>>(x, ln1_g, ln1_b, xn16, 1);

    // qkv GEMM: [25088,256] x [256,768]
    gemm_wmma<<<dim3(768 / 64, TOK / 64), 128, 0, stream>>>(
        xn16, wqkv_t, qkv_b, nullptr, qkv16, TOK, 768, 256, GM_F16);

    split_qkv<<<TOK, 256, 0, stream>>>(qkv16, q16, k16, v16, v32);
    win_mean<<<NB * P2, 256, 0, stream>>>(q16, k16, qwin, kwin);
    routing_topk<<<NB * P2, 64, 0, stream>>>(qwin, kwin, ridx);

    // attention (WMMA flash) + lepe conv
    bra_attention<<<NB * P2 * HEADS, 128, 0, stream>>>(q16, k16, v16, ridx, attn16);
    lepe_conv<<<TOK, 256, 0, stream>>>(v32, lepe_w, lepe_b, lepe32);
    add_lepe_reorder<<<TOK, 256, 0, stream>>>(attn16, lepe32, a16);

    // wo GEMM + residual(x) -> x1
    gemm_wmma<<<dim3(256 / 64, TOK / 64), 128, 0, stream>>>(
        a16, wo_t, wo_b, x, x1, TOK, 256, 256, GM_F32_RES);

    // LN2 + MLP
    ln_kernel<<<TOK, 256, 0, stream>>>(x1, ln2_g, ln2_b, h16, 0);
    gemm_wmma<<<dim3(1024 / 64, TOK / 64), 128, 0, stream>>>(
        h16, w1_t, mlp_b1, nullptr, hid16, TOK, 1024, 256, GM_F16_GELU);
    gemm_wmma<<<dim3(256 / 64, TOK / 64), 128, 0, stream>>>(
        hid16, w2_t, mlp_b2, x1, (float*)d_out, TOK, 256, 1024, GM_F32_RES);
}
